// Net_18124761989248
// MI455X (gfx1250) — matrix operand order repaired and checked by reading, not compiled, not run on hardware
//
#include <hip/hip_runtime.h>
#include <math.h>

typedef __bf16 bf16;
typedef float v8f __attribute__((ext_vector_type(8)));
typedef bf16 bf16x16 __attribute__((ext_vector_type(16)));

#define FLAG_BETA 1
#define FLAG_RELU 2
#define WPB 4   // waves (16x16 N-tiles) per block

// ------------------------------------------------------------------ GEMM ----
// C[b] (f32 MxN) = A[b] (bf16 MxK, row-major) @ B[b] (bf16 KxN) where B is
// supplied transposed (BT: NxK row-major).  One block = 4 waves covering a
// 16x64 output patch; the A tile is staged to LDS once and shared by all 4
// waves; each wave stages its own B tile.  All lda/ldbt in this net are even
// and k0 % 32 == 0, so interior tiles use aligned b32 staging loads.
// Epilogue: +C (beta), /rowdiv, +bias, relu.
struct GemmP {
  const bf16* A; const bf16* BT; float* C;
  long long sA, sB, sC, sRD;
  int lda, ldbt, ldc, M, N, K;
  const float* bias; const float* rowdiv;
  int flags;
};

__global__ __launch_bounds__(128) void k_gemm(GemmP p) {
  const int tid = threadIdx.x;
  const int lane = tid & 31;
  const int w = tid >> 5;
  const int ntn = (p.N + 15) >> 4;
  const int ngrp = (ntn + WPB - 1) / WPB;
  const int tm = blockIdx.x / ngrp;
  const int tg = blockIdx.x % ngrp;
  const int b = blockIdx.y;
  const bf16* A  = p.A  + (long long)b * p.sA;
  const bf16* BT = p.BT + (long long)b * p.sB;
  float*      C  = p.C  + (long long)b * p.sC;
  const int m0 = tm * 16;
  const int n0 = (tg * WPB + w) * 16;
  __shared__ unsigned shA32[16][18];           // 16 x 32 bf16, row stride 36
  __shared__ unsigned shB32[WPB][16][18];
  bf16* shA = (bf16*)shA32;
  bf16* shB = (bf16*)shB32[w];
  const int row = lane & 15, hi = lane >> 4;
  v8f acc = (v8f)0.0f;
  for (int k0 = 0; k0 < p.K; k0 += 32) {
    // ---- stage A tile (cooperative across the whole block) ----
    if (m0 + 16 <= p.M && k0 + 32 <= p.K) {
      #pragma unroll
      for (int j = tid; j < 256; j += 128) {   // 256 dwords, aligned fast path
        int r = j >> 4, cu = j & 15;
        shA32[r][cu] = *(const unsigned*)(A + (long long)(m0 + r) * p.lda + k0 + cu * 2);
      }
    } else {
      for (int j = tid; j < 512; j += 128) {   // zero-padded edge path
        int r = j >> 5, c = j & 31;
        int ar = m0 + r, ac = k0 + c;
        shA[r * 36 + c] = (ar < p.M && ac < p.K) ? A[(long long)ar * p.lda + ac] : (bf16)0.0f;
      }
    }
    // ---- stage this wave's B tile ----
    if (n0 + 16 <= p.N && k0 + 32 <= p.K) {
      #pragma unroll
      for (int i = 0; i < 8; i++) {            // 8 aligned dwords per lane
        int cu = hi * 8 + i;
        shB32[w][row][cu] = *(const unsigned*)(BT + (long long)(n0 + row) * p.ldbt + k0 + cu * 2);
      }
    } else {
      #pragma unroll
      for (int t = 0; t < 16; t++) {
        int br = n0 + t, bc = k0 + lane;
        shB[t * 36 + lane] = (br < p.N && bc < p.K) ? BT[(long long)br * p.ldbt + bc] : (bf16)0.0f;
      }
    }
    if (k0 + 32 < p.K && m0 + row < p.M)       // prefetch next A K-slice
      __builtin_prefetch(A + (long long)(m0 + row) * p.lda + k0 + 32 + hi * 16, 0, 1);
    __syncthreads();
    // ---- fragments (CDNA5 bf16 16x16x32 VGPR layouts) ----
    bf16x16 fa, fb;
    const int ka = hi * 8;
    #pragma unroll
    for (int i = 0; i < 8; i++) { fa[i] = shA[row * 36 + ka + i]; fa[8 + i] = shA[row * 36 + 16 + ka + i]; }
    const int kb = hi * 8;
    #pragma unroll
    for (int i = 0; i < 8; i++) { fb[i] = shB[row * 36 + kb + i]; fb[8 + i] = shB[row * 36 + 16 + kb + i]; }
    acc = __builtin_amdgcn_wmma_f32_16x16x32_bf16(false, fa, false, fb, (short)0, acc, false, false);
    __syncthreads();
  }
  #pragma unroll
  for (int v = 0; v < 8; v++) {
    int r = m0 + v + hi * 8, c = n0 + row;
    if (r < p.M && c < p.N) {
      long long ci = (long long)r * p.ldc + c;
      float val = acc[v];
      if (p.flags & FLAG_BETA) val += C[ci];
      if (p.rowdiv) val /= p.rowdiv[(long long)b * p.sRD + r];
      if (p.bias) val += p.bias[c];
      if (p.flags & FLAG_RELU) val = fmaxf(val, 0.0f);
      C[ci] = val;
    }
  }
}

// ------------------------------------------------------------ small kernels -
struct ConvP { const float* w[8]; const float* b[8]; };

// banded conv -> strict-upper adjacency entries (U must be pre-zeroed)
__global__ void k_band(const float* xin, ConvP cp, float* U, int ch) {
  int b = blockIdx.x, i = blockIdx.y, t = threadIdx.x;
  int k = i + 2, L = 128 - k + 1;
  if (t >= L) return;
  const float* w = cp.w[i];
  float s = cp.b[i][0];
  for (int j = 0; j < k; j++) s += xin[((long long)b * 128 + (t + j)) * 2 + ch] * w[j];
  s = fmaxf(s, 0.0f);
  U[((long long)b * 128 + t) * 128 + (t + i + 1)] = s;
}

__global__ void k_sym(const float* U, float* ADJ, long long total) {
  long long i = blockIdx.x * (long long)blockDim.x + threadIdx.x;
  if (i >= total) return;
  int c = (int)(i % 128); long long t = i / 128;
  int r = (int)(t % 128); long long b = t / 128;
  ADJ[i] = U[(b * 128 + r) * 128 + c] + U[(b * 128 + c) * 128 + r];
}

__global__ void k_deg(const float* adj, float* deg, int Nn) {
  int b = blockIdx.x, r = threadIdx.x;
  if (r >= Nn) return;
  const float* row = adj + ((long long)b * Nn + r) * Nn;
  float s = 0.0f;
  for (int c = 0; c < Nn; c++) s += row[c];
  deg[(long long)b * Nn + r] = fmaxf(s, 1.0f);
}

// f32 -> bf16 (row-major copy + optional transposed copy)
__global__ void k_cvt(const float* in, bf16* out, bf16* outT, long long total, int R, int C) {
  long long i = blockIdx.x * (long long)blockDim.x + threadIdx.x;
  if (i >= total) return;
  int c = (int)(i % C); long long t = i / C;
  int r = (int)(t % R); long long b = t / R;
  bf16 v = (bf16)in[i];
  if (out)  out[i] = v;
  if (outT) outT[(b * C + c) * (long long)R + r] = v;
}

__global__ void k_cvt_off(const float* in, bf16* out, int rows, int C, int ldo) {
  int i = blockIdx.x * blockDim.x + threadIdx.x;
  if (i >= rows * C) return;
  int r = i / C, c = i % C;
  out[(long long)r * ldo + c] = (bf16)in[i];
}

__global__ void k_wT(const float* in, bf16* out, int R, int C) {
  int i = blockIdx.x * blockDim.x + threadIdx.x;
  if (i >= R * C) return;
  int r = i / C, c = i % C;
  out[(long long)c * R + r] = (bf16)in[i];
}

__global__ void k_xiq(const float* xin, bf16* x, bf16* xT) {
  int i = blockIdx.x * blockDim.x + threadIdx.x;
  if (i >= 512 * 128 * 2) return;
  int b = i / 256, r = i % 256, n = r / 2, c = r % 2;
  float v = xin[i] * 10.0f;
  x[i] = (bf16)v;
  xT[(long long)b * 256 + c * 128 + n] = (bf16)v;
}

__global__ void k_bnstats(const float* R, int rows, int F, float* mv) {
  int f = blockIdx.x;
  __shared__ float ss[256], sq[256];
  float s = 0.0f, q = 0.0f;
  for (int r = threadIdx.x; r < rows; r += 256) {
    float v = R[(long long)r * F + f];
    s += v; q += v * v;
  }
  ss[threadIdx.x] = s; sq[threadIdx.x] = q;
  __syncthreads();
  for (int st = 128; st > 0; st >>= 1) {
    if (threadIdx.x < st) { ss[threadIdx.x] += ss[threadIdx.x + st]; sq[threadIdx.x] += sq[threadIdx.x + st]; }
    __syncthreads();
  }
  if (threadIdx.x == 0) {
    float m = ss[0] / rows;
    mv[f] = m;
    mv[F + f] = sq[0] / rows - m * m;
  }
}

struct BnP {
  const float* R; const float* mv; const float* gamma; const float* beta;
  bf16 *h, *hT, *cat, *catT;
  int Nn, F, catld; long long total;
};

__global__ void k_bnapply(BnP p) {
  long long i = blockIdx.x * (long long)blockDim.x + threadIdx.x;
  if (i >= p.total) return;
  int f = (int)(i % p.F); long long t = i / p.F;
  int n = (int)(t % p.Nn); long long b = t / p.Nn;
  float m = p.mv[f], v = p.mv[p.F + f];
  float val = (p.R[i] - m) * rsqrtf(v + 1e-5f) * p.gamma[f] + p.beta[f];
  bf16 bv = (bf16)val;
  p.h[i] = bv;
  p.hT[(b * p.F + f) * (long long)p.Nn + n] = bv;
  p.cat[(b * p.Nn + n) * (long long)p.catld + f] = bv;
  if (p.catT) p.catT[(b * (long long)p.catld + f) * p.Nn + n] = bv;
}

__global__ void k_softmax(const float* sp, bf16* s, bf16* sT, int Nn, int C, float* ent) {
  int b = blockIdx.x, n = threadIdx.x;
  if (n >= Nn) return;
  const float* row = sp + ((long long)b * Nn + n) * C;
  float mx = row[0];
  for (int c = 1; c < C; c++) mx = fmaxf(mx, row[c]);
  float e[32], sum = 0.0f;
  for (int c = 0; c < C; c++) { e[c] = expf(row[c] - mx); sum += e[c]; }
  float inv = 1.0f / sum, el = 0.0f;
  for (int c = 0; c < C; c++) {
    float v = e[c] * inv;
    s[((long long)b * Nn + n) * C + c] = (bf16)v;
    sT[((long long)b * C + c) * Nn + n] = (bf16)v;
    el += -v * logf(v + 1e-15f);
  }
  if (ent) atomicAdd(ent, el);
}

__global__ void k_link(const float* a, const float* g, long long n, float* acc) {
  __shared__ float sm[256];
  float s = 0.0f;
  for (long long i = blockIdx.x * 256ll + threadIdx.x; i < n; i += (long long)gridDim.x * 256)
  { float d = a[i] - g[i]; s += d * d; }
  sm[threadIdx.x] = s;
  __syncthreads();
  for (int st = 128; st > 0; st >>= 1) {
    if (threadIdx.x < st) sm[threadIdx.x] += sm[threadIdx.x + st];
    __syncthreads();
  }
  if (threadIdx.x == 0) atomicAdd(acc, sm[0]);
}

__global__ void k_mean(const bf16* cat, bf16* out, int Nn, int F) {
  int b = blockIdx.x, f = threadIdx.x;
  if (f >= F) return;
  float s = 0.0f;
  for (int n = 0; n < Nn; n++) s += (float)cat[((long long)b * Nn + n) * F + f];
  out[(long long)b * F + f] = (bf16)(s / Nn);
}

__global__ void k_out(const float* logits, const float* acc, float* out) {
  int b = blockIdx.x;
  if (threadIdx.x != 0) return;
  const float* row = logits + (long long)b * 11;
  float m = row[0];
  for (int j = 1; j < 11; j++) m = fmaxf(m, row[j]);
  float s = 0.0f;
  for (int j = 0; j < 11; j++) s += expf(row[j] - m);
  float lse = m + logf(s);
  for (int j = 0; j < 11; j++) out[(long long)b * 11 + j] = row[j] - lse;
  if (b == 0) {
    float l1 = sqrtf(acc[0]) / (512.0f * 128.0f * 128.0f);
    float l2 = sqrtf(acc[1]) / (512.0f * 32.0f * 32.0f);
    out[512 * 11]     = l1 + l2;
    out[512 * 11 + 1] = acc[2] / (512.0f * 128.0f) + acc[3] / (512.0f * 32.0f);
  }
}

// ------------------------------------------------------------------- host ---
static void gemm(hipStream_t st, const bf16* A, long long sA, int lda,
                 const bf16* BT, long long sB, int ldbt,
                 float* C, long long sC, int ldc,
                 int M, int N, int K, int batch,
                 const float* bias, const float* rowdiv, long long sRD, int flags) {
  GemmP p;
  p.A = A; p.BT = BT; p.C = C;
  p.sA = sA; p.sB = sB; p.sC = sC; p.sRD = sRD;
  p.lda = lda; p.ldbt = ldbt; p.ldc = ldc;
  p.M = M; p.N = N; p.K = K;
  p.bias = bias; p.rowdiv = rowdiv; p.flags = flags;
  int ntn = (N + 15) / 16;
  int ngrp = (ntn + WPB - 1) / WPB;
  dim3 grid((unsigned)(((M + 15) / 16) * ngrp), (unsigned)batch);
  k_gemm<<<grid, 128, 0, st>>>(p);
}

struct LayerW { const float *Wrel, *Wroot, *broot, *gamma, *beta; int a, b; bf16 *WrelT, *WrootT; };
struct GnnW   { LayerW L[6]; const float *linW, *linb; int lina, linc; bf16* linWT; };

extern "C" void kernel_launch(void* const* d_in, const int* in_sizes, int n_in,
                              void* d_out, int out_size, void* d_ws, size_t ws_size,
                              hipStream_t stream) {
  (void)in_sizes; (void)n_in; (void)out_size; (void)ws_size;
  char* ws = (char*)d_ws;
  size_t off = 0;
  auto alloc = [&](size_t bytes) -> void* {
    off = (off + 255) & ~(size_t)255;
    void* p = ws + off;
    off += bytes;
    return p;
  };
  auto wT = [&](const float* W, int R, int C) -> bf16* {
    bf16* out = (bf16*)alloc((size_t)R * C * 2);
    int n = R * C;
    k_wT<<<(n + 255) / 256, 256, 0, stream>>>(W, out, R, C);
    return out;
  };

  // ---- parse inputs (setup_inputs dict insertion order, flattened DFS) ----
  int ii = 0;
  auto nf = [&]() { return (const float*)d_in[ii++]; };
  const float* xin = nf();
  ConvP cp;
  for (int i = 0; i < 8; i++) { cp.w[i] = nf(); cp.b[i] = nf(); }
  auto readGnn = [&](int cin, int cout, bool hasLin) {
    GnnW g{};
    int dims[7] = {cin, 64, 64, 64, 64, 64, cout};
    for (int l = 0; l < 6; l++) {
      g.L[l].a = dims[l]; g.L[l].b = dims[l + 1];
      g.L[l].Wrel  = nf(); g.L[l].Wroot = nf(); g.L[l].broot = nf();
      g.L[l].gamma = nf(); g.L[l].beta  = nf();
      g.L[l].WrelT  = wT(g.L[l].Wrel,  dims[l], dims[l + 1]);
      g.L[l].WrootT = wT(g.L[l].Wroot, dims[l], dims[l + 1]);
    }
    if (hasLin) {
      g.lina = 5 * 64 + cout; g.linc = cout;
      g.linW = nf(); g.linb = nf();
      g.linWT = wT(g.linW, g.lina, cout);
    }
    return g;
  };
  GnnW g1p = readGnn(2, 32, true);
  GnnW g1e = readGnn(2, 64, false);
  GnnW g2p = readGnn(384, 8, true);
  GnnW g2e = readGnn(384, 64, false);
  GnnW g3e = readGnn(384, 64, false);
  const float* lin1W = nf(); const float* lin1b = nf();
  const float* lin2W = nf(); const float* lin2b = nf();
  const float* lin3W = nf(); const float* lin3b = nf();
  const float* lin4W = nf(); const float* lin4b = nf();
  bf16* lin1WT = wT(lin1W, 384, 192);
  bf16* lin2WT = wT(lin2W, 192, 64);
  bf16* lin3WT = wT(lin3W, 128, 64);
  bf16* lin4WT = wT(lin4W, 64, 11);

  // ---- scratch arena ----
  bf16* XIQ   = (bf16*)alloc(512ull * 128 * 2 * 2);
  bf16* XIQT  = (bf16*)alloc(512ull * 128 * 2 * 2);
  float* ADJU = (float*)alloc(512ull * 128 * 128 * 4);   // also G (s s^T)
  float* ADJ  = (float*)alloc(512ull * 128 * 128 * 4);
  bf16* ADJB  = (bf16*)alloc(512ull * 128 * 128 * 2);
  float* DEG  = (float*)alloc(512ull * 128 * 4);
  float* AGGF = (float*)alloc(512ull * 12288 * 4);       // also R, x_p (f32)
  bf16* AGGB  = (bf16*)alloc(512ull * 12288 * 2);
  bf16* HBF[2]; bf16* HT[2];
  for (int i = 0; i < 2; i++) {
    HBF[i] = (bf16*)alloc(512ull * 128 * 64 * 2);
    HT[i]  = (bf16*)alloc(512ull * 64 * 128 * 2);
  }
  float* MV   = (float*)alloc(128 * 4);
  bf16* CAT   = (bf16*)alloc(512ull * 128 * 384 * 2);
  bf16* CATT  = (bf16*)alloc(512ull * 384 * 128 * 2);
  float* SPRE = (float*)alloc(512ull * 128 * 32 * 4);
  bf16* SB    = (bf16*)alloc(512ull * 128 * 32 * 2);
  bf16* ST    = (bf16*)alloc(512ull * 32 * 128 * 2);
  float* TF   = (float*)alloc(512ull * 128 * 32 * 4);
  bf16* TB    = (bf16*)alloc(512ull * 128 * 32 * 2);
  bf16* TT    = (bf16*)alloc(512ull * 32 * 128 * 2);
  float* ADJP   = (float*)alloc(512ull * 32 * 32 * 4);
  bf16* ADJPB   = (bf16*)alloc(512ull * 32 * 32 * 2);
  float* ADJP2  = (float*)alloc(512ull * 8 * 8 * 4);
  bf16* ADJP2B  = (bf16*)alloc(512ull * 8 * 8 * 2);
  bf16* XN    = (bf16*)alloc(512ull * 32 * 384 * 2);
  bf16* XNT   = (bf16*)alloc(512ull * 384 * 32 * 2);
  bf16* XN2   = (bf16*)alloc(512ull * 8 * 384 * 2);
  bf16* XN2T  = (bf16*)alloc(512ull * 384 * 8 * 2);
  bf16* MEANX = (bf16*)alloc(512ull * 384 * 2);
  float* L1F  = (float*)alloc(512ull * 192 * 4);
  bf16* L1B   = (bf16*)alloc(512ull * 192 * 2);
  float* L2F  = (float*)alloc(512ull * 64 * 4);
  bf16* HCAT  = (bf16*)alloc(512ull * 128 * 2);
  float* H3F  = (float*)alloc(512ull * 64 * 4);
  bf16* H3B   = (bf16*)alloc(512ull * 64 * 2);
  float* LOGITS = (float*)alloc(512ull * 11 * 4);
  float* ACC  = (float*)alloc(4 * 4);
  float* RF = AGGF;     // agg f32 is dead by the time R / x_p are produced
  float* XP = AGGF;
  float* G  = ADJU;     // U is dead after symmetrization

  auto cvt = [&](const float* in, bf16* out, bf16* outT, int batch, int R, int C) {
    long long tot = (long long)batch * R * C;
    k_cvt<<<(unsigned)((tot + 255) / 256), 256, 0, stream>>>(in, out, outT, tot, R, C);
  };

  auto run_gnn = [&](const GnnW& g, const bf16* x, const bf16* xT, int Nn, int Fin,
                     const bf16* adjb, bool wantCatT) {
    const bf16* hin = x; const bf16* hinT = xT;
    int fin = Fin, coloff = 0, db = 0;
    for (int l = 0; l < 6; l++) {
      int Fo = g.L[l].b;
      // agg = (adj @ h) / deg
      gemm(stream, adjb, (long long)Nn * Nn, Nn, hinT, (long long)fin * Nn, Nn,
           AGGF, (long long)Nn * fin, fin, Nn, fin, Nn, 512, nullptr, DEG, Nn, 0);
      cvt(AGGF, AGGB, nullptr, 512, Nn, fin);
      // pre = agg @ Wrel + h @ Wroot + broot ; relu
      gemm(stream, AGGB, (long long)Nn * fin, fin, g.L[l].WrelT, 0, fin,
           RF, (long long)Nn * Fo, Fo, Nn, Fo, fin, 512, nullptr, nullptr, 0, 0);
      gemm(stream, hin, (long long)Nn * fin, fin, g.L[l].WrootT, 0, fin,
           RF, (long long)Nn * Fo, Fo, Nn, Fo, fin, 512, g.L[l].broot, nullptr, 0,
           FLAG_BETA | FLAG_RELU);
      // batchnorm over all B*N rows
      k_bnstats<<<Fo, 256, 0, stream>>>(RF, 512 * Nn, Fo, MV);
      BnP bp;
      bp.R = RF; bp.mv = MV; bp.gamma = g.L[l].gamma; bp.beta = g.L[l].beta;
      bp.h = HBF[db]; bp.hT = HT[db];
      bp.cat = CAT + coloff;
      bp.catT = wantCatT ? (CATT + (long long)coloff * Nn) : nullptr;
      bp.Nn = Nn; bp.F = Fo; bp.catld = 384;
      bp.total = (long long)512 * Nn * Fo;
      k_bnapply<<<(unsigned)((bp.total + 255) / 256), 256, 0, stream>>>(bp);
      hin = HBF[db]; hinT = HT[db]; fin = Fo; coloff += Fo; db ^= 1;
    }
  };

  auto pool_s = [&](const GnnW& g, int Nn, int Kc, float* entacc) {
    gemm(stream, CAT, (long long)Nn * 384, 384, g.linWT, 0, g.lina,
         SPRE, (long long)Nn * Kc, Kc, Nn, Kc, g.lina, 512, g.linb, nullptr, 0, FLAG_RELU);
    k_softmax<<<512, Nn, 0, stream>>>(SPRE, SB, ST, Nn, Kc, entacc);
  };

  auto diffpool = [&](int Nn, int Kc, const bf16* adjb_cur, const float* adjsrcF,
                      float* adjF_next, bf16* adjB_next, bf16* xnext, bf16* xnextT,
                      float* linkacc) {
    // x_p = s^T x
    gemm(stream, ST, (long long)Kc * Nn, Nn, CATT, (long long)384 * Nn, Nn,
         XP, (long long)Kc * 384, 384, Kc, 384, Nn, 512, nullptr, nullptr, 0, 0);
    cvt(XP, xnext, xnextT, 512, Kc, 384);
    // link loss (q branch only): ||adj - s s^T||_F
    if (linkacc) {
      gemm(stream, SB, (long long)Nn * Kc, Kc, SB, (long long)Nn * Kc, Kc,
           G, (long long)Nn * Nn, Nn, Nn, Nn, Kc, 512, nullptr, nullptr, 0, 0);
      long long tot = 512ll * Nn * Nn;
      k_link<<<2048, 256, 0, stream>>>(adjsrcF, G, tot, linkacc);
    }
    // t = adj @ s ; adj_p = s^T t
    gemm(stream, adjb_cur, (long long)Nn * Nn, Nn, ST, (long long)Kc * Nn, Nn,
         TF, (long long)Nn * Kc, Kc, Nn, Kc, Nn, 512, nullptr, nullptr, 0, 0);
    cvt(TF, TB, TT, 512, Nn, Kc);
    gemm(stream, ST, (long long)Kc * Nn, Nn, TT, (long long)Kc * Nn, Nn,
         adjF_next, (long long)Kc * Kc, Kc, Kc, Kc, Nn, 512, nullptr, nullptr, 0, 0);
    k_deg<<<512, Kc, 0, stream>>>(adjF_next, DEG, Kc);
    cvt(adjF_next, adjB_next, nullptr, 512, Kc, Kc);
  };

  // ---- pipeline ----
  hipMemsetAsync(ACC, 0, 16, stream);
  k_xiq<<<(512 * 128 * 2 + 255) / 256, 256, 0, stream>>>(xin, XIQ, XIQT);

  for (int br = 0; br < 2; br++) {
    bool q = (br == 1);
    // adjacency from channel br
    hipMemsetAsync(ADJU, 0, 512ull * 128 * 128 * 4, stream);
    k_band<<<dim3(512, 8), 128, 0, stream>>>(xin, cp, ADJU, br);
    { long long tot = 512ll * 128 * 128;
      k_sym<<<(unsigned)((tot + 255) / 256), 256, 0, stream>>>(ADJU, ADJ, tot); }
    cvt(ADJ, ADJB, nullptr, 512, 128, 128);
    k_deg<<<512, 128, 0, stream>>>(ADJ, DEG, 128);
    // stage 1: N=128
    run_gnn(g1p, XIQ, XIQT, 128, 2, ADJB, false);
    pool_s(g1p, 128, 32, q ? ACC + 2 : nullptr);
    run_gnn(g1e, XIQ, XIQT, 128, 2, ADJB, true);
    diffpool(128, 32, ADJB, ADJ, ADJP, ADJPB, XN, XNT, q ? ACC + 0 : nullptr);
    // stage 2: N=32
    run_gnn(g2p, XN, XNT, 32, 384, ADJPB, false);
    pool_s(g2p, 32, 8, q ? ACC + 3 : nullptr);
    run_gnn(g2e, XN, XNT, 32, 384, ADJPB, true);
    diffpool(32, 8, ADJPB, ADJP, ADJP2, ADJP2B, XN2, XN2T, q ? ACC + 1 : nullptr);
    // stage 3: N=8
    run_gnn(g3e, XN2, XN2T, 8, 384, ADJP2B, false);
    k_mean<<<512, 384, 0, stream>>>(CAT, MEANX, 8, 384);
    // per-branch head: lin1 relu, lin2 relu -> half of HCAT
    gemm(stream, MEANX, 0, 384, lin1WT, 0, 384, L1F, 0, 192,
         512, 192, 384, 1, lin1b, nullptr, 0, FLAG_RELU);
    cvt(L1F, L1B, nullptr, 1, 512, 192);
    gemm(stream, L1B, 0, 192, lin2WT, 0, 192, L2F, 0, 64,
         512, 64, 192, 1, lin2b, nullptr, 0, FLAG_RELU);
    k_cvt_off<<<(512 * 64 + 255) / 256, 256, 0, stream>>>(L2F, HCAT + br * 64, 512, 64, 128);
  }

  // shared head: lin3 relu, lin4, log_softmax (+ losses)
  gemm(stream, HCAT, 0, 128, lin3WT, 0, 128, H3F, 0, 64,
       512, 64, 128, 1, lin3b, nullptr, 0, FLAG_RELU);
  cvt(H3F, H3B, nullptr, 1, 512, 64);
  gemm(stream, H3B, 0, 64, lin4WT, 0, 64, LOGITS, 0, 11,
       512, 11, 64, 1, lin4b, nullptr, 0, 0);
  k_out<<<512, 32, 0, stream>>>(LOGITS, ACC, (float*)d_out);
}
